// MultiAttackTDATransformer_4114578669662
// MI455X (gfx1250) — compile-verified, matrix operand order checked by reading, not run on hardware
//
#include <hip/hip_runtime.h>
#include <hip/hip_bf16.h>
#include <cmath>

typedef __attribute__((ext_vector_type(16))) _Float16 v16h;
typedef __attribute__((ext_vector_type(8)))  _Float16 v8h;
typedef __attribute__((ext_vector_type(8)))  float    v8f;

#define BNSCALE 0.9999950000374997f  /* 1/sqrt(1+1e-5) */

// ---------------------------------------------------------------------------
// Prep kernels (tiny, run once per launch)
// ---------------------------------------------------------------------------
__global__ void k_xpad(const float* __restrict__ x, _Float16* __restrict__ x16, int B) {
  int i = blockIdx.x * blockDim.x + threadIdx.x;
  if (i >= B * 96) return;
  int row = i / 96, k = i - row * 96;
  x16[i] = (k < 80) ? (_Float16)x[(size_t)row * 80 + k] : (_Float16)0.0f;
}

__global__ void k_wenc(const float* __restrict__ enc_w, _Float16* __restrict__ w16) {
  int i = blockIdx.x * blockDim.x + threadIdx.x;
  if (i >= 512 * 96) return;
  int n = i / 96, k = i - n * 96;
  w16[i] = (k < 80) ? (_Float16)enc_w[(size_t)n * 80 + k] : (_Float16)0.0f;
}

__global__ void k_cvt(const float* __restrict__ src, _Float16* __restrict__ dst, int n) {
  int i = blockIdx.x * blockDim.x + threadIdx.x;
  if (i < n) dst[i] = (_Float16)src[i];
}

__global__ void k_hbias(const float* __restrict__ b0, const float* __restrict__ b1,
                        const float* __restrict__ b2, float* __restrict__ dst) {
  int i = blockIdx.x * blockDim.x + threadIdx.x;
  if (i >= 384) return;
  dst[i] = (i < 128) ? b0[i] : (i < 256) ? b1[i - 128] : b2[i - 256];
}

// C = Wo @ Wv  (fused S=1 attention); job0 = kc, jobs1..6 = transformer layers
__global__ void k_combine(const float* __restrict__ kc_wi, const float* __restrict__ kc_wo,
                          const float* __restrict__ tf_wi, const float* __restrict__ tf_wo,
                          _Float16* __restrict__ C16) {
  int job = blockIdx.y;
  int e = blockIdx.x * blockDim.x + threadIdx.x;       // 0..65535
  int n = e >> 8, k = e & 255;
  const float *Wo, *Wv;
  if (job == 0) { Wo = kc_wo; Wv = kc_wi + 512 * 256; }
  else { int i = job - 1; Wo = tf_wo + (size_t)i * 65536; Wv = tf_wi + (size_t)i * 196608 + 131072; }
  float acc = 0.f;
  for (int j = 0; j < 256; ++j) acc += Wo[n * 256 + j] * Wv[j * 256 + k];
  C16[(size_t)job * 65536 + e] = (_Float16)acc;
}

// d = Wo @ bv + bo
__global__ void k_cbias(const float* __restrict__ kc_bi, const float* __restrict__ kc_bo,
                        const float* __restrict__ kc_wo,
                        const float* __restrict__ tf_bi, const float* __restrict__ tf_bo,
                        const float* __restrict__ tf_wo, float* __restrict__ DSA) {
  int job = blockIdx.x, n = threadIdx.x;
  const float *Wo, *bv, *bo;
  if (job == 0) { Wo = kc_wo; bv = kc_bi + 512; bo = kc_bo; }
  else { int i = job - 1; Wo = tf_wo + (size_t)i * 65536; bv = tf_bi + (size_t)i * 768 + 512; bo = tf_bo + (size_t)i * 256; }
  float acc = bo[n];
  for (int j = 0; j < 256; ++j) acc += Wo[n * 256 + j] * bv[j];
  DSA[job * 256 + n] = acc;
}

// PA stage1: V6 = Wv @ proj_w (256x6), t = Wv @ proj_b + bv
__global__ void k_pa1(const float* __restrict__ pt_pw, const float* __restrict__ pt_pb,
                      const float* __restrict__ pt_wi, const float* __restrict__ pt_bi,
                      const float* __restrict__ pg_pw, const float* __restrict__ pg_pb,
                      const float* __restrict__ pg_wi, const float* __restrict__ pg_bi,
                      float* __restrict__ V6, float* __restrict__ TV) {
  int j = blockIdx.x, p = blockIdx.y, v = threadIdx.x;
  const float* Wv = (p ? pg_wi : pt_wi) + 512 * 256;
  const float* pw = p ? pg_pw : pt_pw;
  const float* pb = p ? pg_pb : pt_pb;
  const float* bv = (p ? pg_bi : pt_bi) + 512;
  if (j < 6) {
    float acc = 0.f;
    for (int e = 0; e < 256; ++e) acc += Wv[v * 256 + e] * pw[e * 6 + j];
    V6[p * 1536 + v * 6 + j] = acc;
  } else {
    float acc = bv[v];
    for (int e = 0; e < 256; ++e) acc += Wv[v * 256 + e] * pb[e];
    TV[p * 256 + v] = acc;
  }
}

// PA stage2: M6 = Wo @ V6 (256x6), dpa = Wo @ t + bo
__global__ void k_pa2(const float* __restrict__ pt_wo, const float* __restrict__ pt_bo,
                      const float* __restrict__ pg_wo, const float* __restrict__ pg_bo,
                      const float* __restrict__ V6, const float* __restrict__ TV,
                      float* __restrict__ M6, float* __restrict__ DPA) {
  int j = blockIdx.x, p = blockIdx.y, n = threadIdx.x;
  const float* Wo = p ? pg_wo : pt_wo;
  const float* bo = p ? pg_bo : pt_bo;
  if (j < 6) {
    float acc = 0.f;
    for (int v = 0; v < 256; ++v) acc += Wo[n * 256 + v] * V6[p * 1536 + v * 6 + j];
    M6[p * 1536 + n * 6 + j] = acc;
  } else {
    float acc = bo[n];
    for (int v = 0; v < 256; ++v) acc += Wo[n * 256 + v] * TV[p * 256 + v];
    DPA[p * 256 + n] = acc;
  }
}

// ---------------------------------------------------------------------------
// TDA: per-sample 20x20 distance-matrix summaries
// ---------------------------------------------------------------------------
__device__ __forceinline__ void tda6(const float* fi, int temporal, float* out) {
  float nn[20], sum = 0.f, sq = 0.f, dmax = 0.f;
  for (int i = 0; i < 20; ++i) nn[i] = 1e30f;
  for (int i = 0; i < 20; ++i)
    for (int j = 0; j < 20; ++j) {
      if (i == j) continue;
      float dx, dy;
      if (temporal) { dx = (float)(i - j); dy = fi[i] - fi[j]; }
      else          { float d0 = fi[i] - fi[j]; dx = d0; dy = 0.1f * d0; }
      float d = sqrtf(dx * dx + dy * dy + 1e-12f);
      sum += d; sq += d * d;
      if (d < nn[i]) nn[i] = d;
      if (d > dmax) dmax = d;
    }
  const float cnt = 380.0f;
  float mean = sum / cnt;
  float var = fmaxf(sq / cnt - mean * mean, 0.0f);
  float nm = 0.f;
  for (int i = 0; i < 20; ++i) nm += nn[i];
  nm *= (1.0f / 20.0f);
  float nv = 0.f, nmax = 0.f;
  for (int i = 0; i < 20; ++i) { float t = nn[i] - nm; nv += t * t; nmax = fmaxf(nmax, nn[i]); }
  nv *= (1.0f / 20.0f);
  out[0] = nm; out[1] = sqrtf(nv); out[2] = nmax;
  out[3] = mean; out[4] = sqrtf(var + 1e-12f); out[5] = dmax;
}

__global__ void k_tda(const float* __restrict__ x, float* __restrict__ tf, int B) {
  int s = blockIdx.x * blockDim.x + threadIdx.x;
  if (s >= B) return;
  float fi[20];
  for (int i = 0; i < 20; ++i) fi[i] = x[(size_t)s * 80 + i];
  tda6(fi, 1, tf + (size_t)s * 12);
  tda6(fi, 0, tf + (size_t)s * 12 + 6);
}

// ---------------------------------------------------------------------------
// Generic WMMA GEMM: 32 batch rows per block (two 16-row groups sharing each
// B fragment), NWAVES waves x 64 cols.  A[M,K] f16 compact, W[N,K] f16.
// Epilogues:
//  0 enc (bn-relu)  1 fus (bn-relu)  2 kc+pa (R32+f16)  3 sa+res+LN
//  4 gelu           5 ffn2+res+LN (+feat)               6 relu384
// ---------------------------------------------------------------------------
template<int NWAVES, int EP>
__global__ __launch_bounds__(NWAVES * 32)
void gemm32(const _Float16* __restrict__ A, const _Float16* __restrict__ W, int K,
            const float* __restrict__ bias, const float* __restrict__ g,
            const float* __restrict__ beta, float* __restrict__ R32,
            _Float16* __restrict__ O16, int ldo, float* __restrict__ feat,
            const float* __restrict__ tf, const float* __restrict__ M6t,
            const float* __restrict__ M6g, const float* __restrict__ dpt,
            const float* __restrict__ dpg) {
  extern __shared__ __align__(16) char smem[];
  _Float16* As = (_Float16*)smem;
  __shared__ float rsum[32], rsq[32];

  const int tid  = threadIdx.x;
  const int w    = tid >> 5;
  const int lane = tid & 31;
  const int half = lane >> 4;
  const int r    = lane & 15;
  const long m0  = (long)blockIdx.x * 32;

  // Stage the 32xK activation tile through LDS (contiguous region)
  {
    const v8h* src = (const v8h*)(A + m0 * K);
    v8h* dst = (v8h*)As;
    for (int i = tid; i < 4 * K; i += NWAVES * 32) dst[i] = src[i];
  }
  __syncthreads();

  v8f acc[2][4];
  #pragma unroll
  for (int gq = 0; gq < 2; ++gq)
    #pragma unroll
    for (int t = 0; t < 4; ++t)
      acc[gq][t] = (v8f){0.f, 0.f, 0.f, 0.f, 0.f, 0.f, 0.f, 0.f};

  const int kst = K >> 5;
  for (int kt = 0; kt < kst; ++kt) {
    const int kb = kt * 32;
    // Two A fragments (row groups 0-15 and 16-31)
    v16h a0, a1;
    {
      const _Float16* ap0 = As + r * K + kb + half * 8;
      const _Float16* ap1 = As + (16 + r) * K + kb + half * 8;
      v8h l0 = *(const v8h*)ap0,  h0 = *(const v8h*)(ap0 + 16);
      v8h l1 = *(const v8h*)ap1,  h1 = *(const v8h*)(ap1 + 16);
      #pragma unroll
      for (int i = 0; i < 8; ++i) {
        a0[i] = l0[i]; a0[i + 8] = h0[i];
        a1[i] = l1[i]; a1[i + 8] = h1[i];
      }
    }
    // Load ALL 4 B fragments first so the scheduler can clause the loads
    v16h bfr[4];
    #pragma unroll
    for (int t = 0; t < 4; ++t) {
      const _Float16* bp = W + (long)(w * 64 + t * 16 + r) * K + kb + half * 8;
      if (t == 0 && kt + 1 < kst) __builtin_prefetch(bp + 32, 0, 3);
      v8h blo = *(const v8h*)bp;
      v8h bhi = *(const v8h*)(bp + 16);
      #pragma unroll
      for (int i = 0; i < 8; ++i) { bfr[t][i] = blo[i]; bfr[t][i + 8] = bhi[i]; }
    }
    // 8 WMMA per k-step: each B fragment reused by both row groups
    #pragma unroll
    for (int t = 0; t < 4; ++t) {
      acc[0][t] = __builtin_amdgcn_wmma_f32_16x16x32_f16(false, a0, false, bfr[t],
                                                         (short)0, acc[0][t], false, false);
      acc[1][t] = __builtin_amdgcn_wmma_f32_16x16x32_f16(false, a1, false, bfr[t],
                                                         (short)0, acc[1][t], false, false);
    }
  }

  // ---------------- epilogues ----------------
  if (EP == 0 || EP == 1) {  // bn(relu(x + b))
    #pragma unroll
    for (int gq = 0; gq < 2; ++gq)
      #pragma unroll
      for (int t = 0; t < 4; ++t) {
        const int col = w * 64 + t * 16 + r;
        #pragma unroll
        for (int v = 0; v < 8; ++v) {
          const int row = gq * 16 + v + 8 * half;
          float x = acc[gq][t][v] + bias[col];
          x = fmaxf(x, 0.f);
          x = x * (g[col] * BNSCALE) + beta[col];
          O16[(size_t)(m0 + row) * ldo + col] = (_Float16)x;
        }
      }
  } else if (EP == 2) {  // c = kc + pa_t + pa_g  -> R32 + f16
    #pragma unroll
    for (int gq = 0; gq < 2; ++gq)
      #pragma unroll
      for (int t = 0; t < 4; ++t) {
        const int col = w * 64 + t * 16 + r;
        #pragma unroll
        for (int v = 0; v < 8; ++v) {
          const int row = gq * 16 + v + 8 * half;
          const float* tfr = tf + (size_t)(m0 + row) * 12;
          float x = acc[gq][t][v] + bias[col] + dpt[col] + dpg[col];
          #pragma unroll
          for (int j = 0; j < 6; ++j)
            x += tfr[j] * M6t[col * 6 + j] + tfr[6 + j] * M6g[col * 6 + j];
          const size_t gi = (size_t)(m0 + row) * 256 + col;
          R32[gi] = x;
          O16[gi] = (_Float16)x;
        }
      }
  } else if (EP == 4) {  // exact gelu
    #pragma unroll
    for (int gq = 0; gq < 2; ++gq)
      #pragma unroll
      for (int t = 0; t < 4; ++t) {
        const int col = w * 64 + t * 16 + r;
        #pragma unroll
        for (int v = 0; v < 8; ++v) {
          const int row = gq * 16 + v + 8 * half;
          float x = acc[gq][t][v] + bias[col];
          x = 0.5f * x * (1.0f + erff(x * 0.70710678118f));
          O16[(size_t)(m0 + row) * ldo + col] = (_Float16)x;
        }
      }
  } else if (EP == 6) {  // relu heads-hidden
    #pragma unroll
    for (int gq = 0; gq < 2; ++gq)
      #pragma unroll
      for (int t = 0; t < 4; ++t) {
        const int col = w * 64 + t * 16 + r;
        #pragma unroll
        for (int v = 0; v < 8; ++v) {
          const int row = gq * 16 + v + 8 * half;
          float x = fmaxf(acc[gq][t][v] + bias[col], 0.f);
          O16[(size_t)(m0 + row) * ldo + col] = (_Float16)x;
        }
      }
  } else {  // EP 3 / 5: residual + LayerNorm (rows normalized across 256 cols)
    #pragma unroll
    for (int gq = 0; gq < 2; ++gq)
      #pragma unroll
      for (int t = 0; t < 4; ++t) {
        const int col = w * 64 + t * 16 + r;
        #pragma unroll
        for (int v = 0; v < 8; ++v) {
          const int row = gq * 16 + v + 8 * half;
          acc[gq][t][v] += bias[col] + R32[(size_t)(m0 + row) * 256 + col];
        }
      }
    if (tid < 32) { rsum[tid] = 0.f; rsq[tid] = 0.f; }
    __syncthreads();
    #pragma unroll
    for (int gq = 0; gq < 2; ++gq)
      #pragma unroll
      for (int v = 0; v < 8; ++v) {
        float s = 0.f, q = 0.f;
        #pragma unroll
        for (int t = 0; t < 4; ++t) { float x = acc[gq][t][v]; s += x; q += x * x; }
        // reduce across the 16 lanes of this half (rows stay within the half)
        for (int off = 1; off < 16; off <<= 1) {
          s += __shfl_xor(s, off, 32);
          q += __shfl_xor(q, off, 32);
        }
        if (r == 0) {  // one lane per (wave, half, group) -> ds_add_f32
          atomicAdd(&rsum[gq * 16 + v + 8 * half], s);
          atomicAdd(&rsq[gq * 16 + v + 8 * half], q);
        }
      }
    __syncthreads();
    #pragma unroll
    for (int gq = 0; gq < 2; ++gq)
      #pragma unroll
      for (int v = 0; v < 8; ++v) {
        const int row = gq * 16 + v + 8 * half;
        const float m = rsum[row] * (1.0f / 256.0f);
        const float var = rsq[row] * (1.0f / 256.0f) - m * m;
        const float rstd = rsqrtf(var + 1e-5f);
        #pragma unroll
        for (int t = 0; t < 4; ++t) {
          const int col = w * 64 + t * 16 + r;
          const size_t gi = (size_t)(m0 + row) * 256 + col;
          float x = (acc[gq][t][v] - m) * rstd * g[col] + beta[col];
          R32[gi] = x;
          O16[gi] = (_Float16)x;
          if (EP == 5) { if (feat) feat[gi] = x; }
        }
      }
  }
}

// ---------------------------------------------------------------------------
// Final tiny head outputs (13 logits per sample)
// ---------------------------------------------------------------------------
__global__ void k_heads(const _Float16* __restrict__ hh,
                        const float* __restrict__ w2b, const float* __restrict__ b2b,
                        const float* __restrict__ w2a, const float* __restrict__ b2a,
                        const float* __restrict__ w2p, const float* __restrict__ b2p,
                        float* __restrict__ out, int B) {
  int idx = blockIdx.x * blockDim.x + threadIdx.x;
  if (idx >= 13 * B) return;
  int s = idx % B, o = idx / B;
  const float *w, *bb;
  int j, base;
  float* dst;
  if (o < 2)      { j = o;     w = w2b; bb = b2b; base = 0;   dst = out + (size_t)s * 2 + j; }
  else if (o < 7) { j = o - 2; w = w2a; bb = b2a; base = 128; dst = out + (size_t)2 * B + (size_t)s * 5 + j; }
  else            { j = o - 7; w = w2p; bb = b2p; base = 256; dst = out + (size_t)7 * B + (size_t)s * 6 + j; }
  const _Float16* h = hh + (size_t)s * 384 + base;
  float acc = bb[j];
  for (int k = 0; k < 128; ++k) acc += (float)h[k] * w[j * 128 + k];
  *dst = acc;
}

// ---------------------------------------------------------------------------
// Host launcher
// ---------------------------------------------------------------------------
extern "C" void kernel_launch(void* const* d_in, const int* in_sizes, int n_in,
                              void* d_out, int out_size, void* d_ws, size_t ws_size,
                              hipStream_t stream) {
  const float* xin      = (const float*)d_in[0];
  const float* enc_w    = (const float*)d_in[1];
  const float* enc_b    = (const float*)d_in[2];
  const float* enc_g    = (const float*)d_in[3];
  const float* enc_beta = (const float*)d_in[4];
  const float* fus_w    = (const float*)d_in[5];
  const float* fus_b    = (const float*)d_in[6];
  const float* fus_g    = (const float*)d_in[7];
  const float* fus_beta = (const float*)d_in[8];
  const float* kc_wi    = (const float*)d_in[9];
  const float* kc_bi    = (const float*)d_in[10];
  const float* kc_wo    = (const float*)d_in[11];
  const float* kc_bo    = (const float*)d_in[12];
  const float* pt_pw = (const float*)d_in[13];
  const float* pt_pb = (const float*)d_in[14];
  const float* pt_wi = (const float*)d_in[15];
  const float* pt_bi = (const float*)d_in[16];
  const float* pt_wo = (const float*)d_in[17];
  const float* pt_bo = (const float*)d_in[18];
  const float* pg_pw = (const float*)d_in[19];
  const float* pg_pb = (const float*)d_in[20];
  const float* pg_wi = (const float*)d_in[21];
  const float* pg_bi = (const float*)d_in[22];
  const float* pg_wo = (const float*)d_in[23];
  const float* pg_bo = (const float*)d_in[24];
  const float* tf_wi = (const float*)d_in[25];
  const float* tf_bi = (const float*)d_in[26];
  const float* tf_wo = (const float*)d_in[27];
  const float* tf_bo = (const float*)d_in[28];
  const float* ln1g  = (const float*)d_in[29];
  const float* ln1b  = (const float*)d_in[30];
  const float* w1    = (const float*)d_in[31];
  const float* b1    = (const float*)d_in[32];
  const float* w2    = (const float*)d_in[33];
  const float* b2    = (const float*)d_in[34];
  const float* ln2g  = (const float*)d_in[35];
  const float* ln2b  = (const float*)d_in[36];
  const float* binw1 = (const float*)d_in[37];
  const float* binb1 = (const float*)d_in[38];
  const float* binw2 = (const float*)d_in[39];
  const float* binb2 = (const float*)d_in[40];
  const float* atkw1 = (const float*)d_in[41];
  const float* atkb1 = (const float*)d_in[42];
  const float* atkw2 = (const float*)d_in[43];
  const float* atkb2 = (const float*)d_in[44];
  const float* phw1  = (const float*)d_in[45];
  const float* phb1  = (const float*)d_in[46];
  const float* phw2  = (const float*)d_in[47];
  const float* phb2  = (const float*)d_in[48];

  const int B = in_sizes[0] / 80;
  float* out = (float*)d_out;

  char* ws = (char*)d_ws;
  size_t off = 0;
  auto alloc = [&](size_t bytes) -> void* {
    void* p = ws + off;
    off = (off + bytes + 255) & ~(size_t)255;
    return p;
  };
  _Float16* U16  = (_Float16*)alloc((size_t)B * 1024 * 2);  // comb / gelu / heads-hidden
  _Float16* A16a = (_Float16*)alloc((size_t)B * 256 * 2);
  _Float16* A16b = (_Float16*)alloc((size_t)B * 256 * 2);
  float*    R32  = (float*)alloc((size_t)B * 256 * 4);      // residual stream
  _Float16* X16  = (_Float16*)alloc((size_t)B * 96 * 2);
  float*    TF   = (float*)alloc((size_t)B * 12 * 4);
  _Float16* WENC = (_Float16*)alloc(512 * 96 * 2);
  _Float16* WFUS = (_Float16*)alloc(256 * 512 * 2);
  _Float16* C16  = (_Float16*)alloc(7 * 65536 * 2);
  float*    DSA  = (float*)alloc(7 * 256 * 4);
  _Float16* W1H  = (_Float16*)alloc((size_t)6 * 262144 * 2);
  _Float16* W2H  = (_Float16*)alloc((size_t)6 * 262144 * 2);
  _Float16* HW   = (_Float16*)alloc(384 * 256 * 2);
  float*    HB   = (float*)alloc(384 * 4);
  float*    V6   = (float*)alloc(2 * 1536 * 4);
  float*    TV   = (float*)alloc(2 * 256 * 4);
  float*    M6   = (float*)alloc(2 * 1536 * 4);
  float*    DPA  = (float*)alloc(2 * 256 * 4);

  // ---- prep ----
  k_xpad<<<(B * 96 + 255) / 256, 256, 0, stream>>>(xin, X16, B);
  k_wenc<<<(512 * 96 + 255) / 256, 256, 0, stream>>>(enc_w, WENC);
  k_cvt<<<(256 * 512 + 255) / 256, 256, 0, stream>>>(fus_w, WFUS, 256 * 512);
  k_cvt<<<(6 * 262144 + 255) / 256, 256, 0, stream>>>(w1, W1H, 6 * 262144);
  k_cvt<<<(6 * 262144 + 255) / 256, 256, 0, stream>>>(w2, W2H, 6 * 262144);
  k_cvt<<<(32768 + 255) / 256, 256, 0, stream>>>(binw1, HW, 32768);
  k_cvt<<<(32768 + 255) / 256, 256, 0, stream>>>(atkw1, HW + 32768, 32768);
  k_cvt<<<(32768 + 255) / 256, 256, 0, stream>>>(phw1, HW + 65536, 32768);
  k_hbias<<<2, 192, 0, stream>>>(binb1, atkb1, phb1, HB);
  k_combine<<<dim3(256, 7), 256, 0, stream>>>(kc_wi, kc_wo, tf_wi, tf_wo, C16);
  k_cbias<<<7, 256, 0, stream>>>(kc_bi, kc_bo, kc_wo, tf_bi, tf_bo, tf_wo, DSA);
  k_pa1<<<dim3(7, 2), 256, 0, stream>>>(pt_pw, pt_pb, pt_wi, pt_bi, pg_pw, pg_pb, pg_wi, pg_bi, V6, TV);
  k_pa2<<<dim3(7, 2), 256, 0, stream>>>(pt_wo, pt_bo, pg_wo, pg_bo, V6, TV, M6, DPA);
  k_tda<<<(B + 255) / 256, 256, 0, stream>>>(xin, TF, B);

  // ---- main pipeline (grid over 32-row batch tiles) ----
  const int nb = B / 32;
  // encoders: x(96) -> comb(512), bn-relu
  gemm32<8, 0><<<nb, 256, 32 * 96 * 2, stream>>>(X16, WENC, 96, enc_b, enc_g, enc_beta,
      nullptr, U16, 512, nullptr, nullptr, nullptr, nullptr, nullptr, nullptr);
  // fusion: comb(512) -> f(256), bn-relu
  gemm32<4, 1><<<nb, 128, 32 * 512 * 2, stream>>>(U16, WFUS, 512, fus_b, fus_g, fus_beta,
      nullptr, A16a, 256, nullptr, nullptr, nullptr, nullptr, nullptr, nullptr);
  // kc attention (fused) + both PA branches -> c
  gemm32<4, 2><<<nb, 128, 32 * 256 * 2, stream>>>(A16a, C16, 256, DSA, nullptr, nullptr,
      R32, A16b, 256, nullptr, TF, M6, M6 + 1536, DPA, DPA + 256);
  // 6 transformer layers
  for (int i = 0; i < 6; ++i) {
    gemm32<4, 3><<<nb, 128, 32 * 256 * 2, stream>>>(A16b, C16 + (size_t)(1 + i) * 65536, 256,
        DSA + (1 + i) * 256, ln1g + (size_t)i * 256, ln1b + (size_t)i * 256,
        R32, A16a, 256, nullptr, nullptr, nullptr, nullptr, nullptr, nullptr);
    gemm32<16, 4><<<nb, 512, 32 * 256 * 2, stream>>>(A16a, W1H + (size_t)i * 262144, 256,
        b1 + (size_t)i * 1024, nullptr, nullptr, nullptr, U16, 1024, nullptr,
        nullptr, nullptr, nullptr, nullptr, nullptr);
    float* featp = (i == 5) ? (out + (size_t)13 * B) : nullptr;
    gemm32<4, 5><<<nb, 128, 32 * 1024 * 2, stream>>>(U16, W2H + (size_t)i * 262144, 1024,
        b2 + (size_t)i * 256, ln2g + (size_t)i * 256, ln2b + (size_t)i * 256,
        R32, A16b, 256, featp, nullptr, nullptr, nullptr, nullptr, nullptr);
  }
  // heads hidden: feat(256) -> relu(384)
  gemm32<6, 6><<<nb, 192, 32 * 256 * 2, stream>>>(A16b, HW, 256, HB, nullptr, nullptr,
      nullptr, U16, 384, nullptr, nullptr, nullptr, nullptr, nullptr, nullptr);
  // final 13 logits
  k_heads<<<(13 * B + 255) / 256, 256, 0, stream>>>(U16, binw2, binb2, atkw2, atkb2,
                                                    phw2, phb2, out, B);
}
